// _EdgeWeightsGraphConvLayer_11957188952551
// MI455X (gfx1250) — matrix operand order repaired and checked by reading, not compiled, not run on hardware
//
#include <hip/hip_runtime.h>

// ---------------------------------------------------------------------------
// out_g = A^T * X_g * W_rel^T + X_g * W_root^T + bias   (A shared by all graphs)
// All matmuls via v_wmma_f32_16x16x32_bf16 (bf16 in, fp32 accumulate).
// Graph-aligned 16x16 tiles (19 rows padded to 32); pad rows computed but never
// stored. fp32->bf16 packing via v_fma_mixlo/mixhi_bf16 (VOP3P ops 62/63).
// ---------------------------------------------------------------------------

typedef __attribute__((ext_vector_type(16))) __bf16 v16bf;
typedef __attribute__((ext_vector_type(8)))  float  v8f;
typedef __attribute__((ext_vector_type(4)))  float  f4;

#define N_EL 19
#define GPW  4                 // graphs per workgroup
#define ROWS (GPW * N_EL)      // 76 valid rows per workgroup
#define ROWP 96                // padded rows (covers g*19 + 31 reads, g<=3)
#define CIN  64
#define COUT 64

union Frag { v16bf v; uint4 q[2]; };

// Two fp32 -> packed bf16x2 in one dword, 2 VALU ops (CDNA5 V_FMA_MIXLO/HI_BF16).
__device__ __forceinline__ unsigned pk2(float lo, float hi) {
    unsigned d;
    asm("v_fma_mixlo_bf16 %0, %1, 1.0, 0\n\t"
        "v_fma_mixhi_bf16 %0, %2, 1.0, 0"
        : "=&v"(d) : "v"(lo), "v"(hi));
    return d;
}
__device__ __forceinline__ unsigned short f2bf(float f) {
    unsigned d;
    asm("v_fma_mixlo_bf16 %0, %1, 1.0, 0" : "=v"(d) : "v"(f));
    return (unsigned short)d;
}

// A-operand fragment (16xK, bf16) from row-major LDS (K pairs contiguous).
// ISA 7.12.2: lanes 0-15 row=l16, K=kk+{0..7,16..23}; lanes 16-31: +8.
__device__ __forceinline__ v16bf load_a(const unsigned short* rowPtr, int kk, int hf8) {
    Frag a;
    a.q[0] = *(const uint4*)(rowPtr + kk + hf8);
    a.q[1] = *(const uint4*)(rowPtr + kk + 16 + hf8);
    return a.v;
}
// B-operand fragment (Kx16, bf16) from [n][k]-major LDS:
// lane n=l16 holds K=kk+{0..7,16..23} (lanes 0-15); lanes 16-31: +8 (matches load_a).
__device__ __forceinline__ v16bf load_b(const unsigned short* colPtr, int kk, int hf16) {
    Frag b;
    b.q[0] = *(const uint4*)(colPtr + kk + (hf16 >> 1));
    b.q[1] = *(const uint4*)(colPtr + kk + 16 + (hf16 >> 1));
    return b.v;
}

#define WMMA(a, b, c) __builtin_amdgcn_wmma_f32_16x16x32_bf16( \
        false, (a), false, (b), (short)0, (c), false, false)

__global__ __launch_bounds__(256) void graphconv_fused_kernel(
    const float* __restrict__ x, const float* __restrict__ ew,
    const float* __restrict__ Wrel, const float* __restrict__ Wroot,
    const float* __restrict__ bias, float* __restrict__ out)
{
    __shared__ unsigned short Xs [ROWP * CIN];       // X block, bf16, row-major
    __shared__ unsigned short WrT[COUT * CIN];       // W_rel  [n][k] bf16
    __shared__ unsigned short WoT[COUT * CIN];       // W_root [n][k] bf16
    __shared__ unsigned short At [32 * 32];          // softplus(A)^T, zero-padded
    __shared__ unsigned short Yt [GPW * COUT * 32];  // Y per graph [g][n][k], K-pad 32
    __shared__ float biasS[COUT];

    const int tid  = threadIdx.x;
    const int lane = tid & 31;
    const int wave = tid >> 5;
    const int hf   = lane >> 4;
    const int l16  = lane & 15;
    const int hf8  = hf * 8, hf16 = hf * 16;

    const long nodeBase = (long)blockIdx.x * ROWS;
    const f4* xg4 = (const f4*)(x + nodeBase * CIN);

    // ---- cooperative staging: coalesced fp32 loads -> packed bf16 LDS ----
    for (int d4 = tid; d4 < (ROWS * CIN) / 4; d4 += 256) {
        f4 v = __builtin_nontemporal_load(xg4 + d4);      // streamed, read once
        uint2 p; p.x = pk2(v[0], v[1]); p.y = pk2(v[2], v[3]);
        *(uint2*)&Xs[d4 * 4] = p;
    }
    for (int i4 = tid; i4 < (COUT * CIN) / 4; i4 += 256) { // natural [n][k] == B layout
        f4 a = *((const f4*)Wrel  + i4);
        f4 b = *((const f4*)Wroot + i4);
        uint2 pa; pa.x = pk2(a[0], a[1]); pa.y = pk2(a[2], a[3]);
        uint2 pb; pb.x = pk2(b[0], b[1]); pb.y = pk2(b[2], b[3]);
        *(uint2*)&WrT[i4 * 4] = pa;
        *(uint2*)&WoT[i4 * 4] = pb;
    }
    // single-pass A^T fill: At[i][j] = (i<19 && j<19) ? softplus(ew[j*19+i]) : 0
    for (int idx = tid; idx < 32 * 32; idx += 256) {
        int i = idx >> 5, j = idx & 31;                   // shift/mask, no division
        float v = 0.f;
        if (i < N_EL && j < N_EL) {
            float w = ew[j * N_EL + i];
            v = fmaxf(w, 0.f) + log1pf(expf(-fabsf(w)));
        }
        At[idx] = f2bf(v);
    }
    // zero only the K in [16,32) half of Yt rows; K in [0,16) is fully written
    // by the mt=0 ds_store_b128 below. (Keeps NaN out of the K-pad.)
    for (int idx = tid; idx < GPW * COUT * 8; idx += 256) {
        int row = idx >> 3, d = idx & 7;
        ((unsigned*)Yt)[row * 16 + 8 + d] = 0;
    }
    if (tid < COUT) biasS[tid] = bias[tid];
    __syncthreads();

    // ---- stage 1: Y_g = X_g @ W_rel^T, graph-aligned tiles, Y stored [n][k] ----
    for (int idx = wave; idx < GPW * 4; idx += 8) {   // (graph, ntile): 16 tiles
        int g = idx >> 2, nt = idx & 3;
        int col = nt * 16 + l16;
        const unsigned short* bcol = &WrT[col * CIN];
        v16bf b0 = load_b(bcol, 0, hf16);
        v16bf b1 = load_b(bcol, 32, hf16);
        const unsigned short* xrow = &Xs[(g * N_EL) * CIN];
        unsigned short* yb = &Yt[(g * COUT + col) * 32];
        { // mt = 0: rows 0..15, all valid -> one packed ds_store_b128
            const unsigned short* ar = xrow + l16 * CIN;
            v16bf a0 = load_a(ar, 0, hf8), a1 = load_a(ar, 32, hf8);
            v8f c = {};
            c = WMMA(a0, b0, c);
            c = WMMA(a1, b1, c);
            uint4 q;
            q.x = pk2(c[0], c[1]); q.y = pk2(c[2], c[3]);
            q.z = pk2(c[4], c[5]); q.w = pk2(c[6], c[7]);
            *(uint4*)(yb + hf8) = q;                   // k = hf8..hf8+7, 16B aligned
        }
        { // mt = 1: only rows 16..18 kept (lanes hf==0, c[0..2])
            const unsigned short* ar = xrow + (16 + l16) * CIN;
            v16bf a0 = load_a(ar, 0, hf8), a1 = load_a(ar, 32, hf8);
            v8f c = {};
            c = WMMA(a0, b0, c);
            c = WMMA(a1, b1, c);
            if (hf == 0) {
                *(unsigned*)(yb + 16) = pk2(c[0], c[1]);
                yb[18] = f2bf(c[2]);
            }
        }
    }
    __syncthreads();

    // ---- stage 2: out_g = bias + A^T @ Y_g + X_g @ W_root^T ----
    for (int idx = wave; idx < GPW * 4; idx += 8) {
        int g = idx >> 2, nt = idx & 3;
        int col = nt * 16 + l16;
        float bv = biasS[col];
        v16bf by  = load_b(&Yt[(g * COUT + col) * 32], 0, hf16);  // K=32 covers K=19
        const unsigned short* wb = &WoT[col * CIN];
        v16bf bw0 = load_b(wb, 0, hf16);
        v16bf bw1 = load_b(wb, 32, hf16);
        const unsigned short* xrow = &Xs[(g * N_EL) * CIN];
        float* og = out + (nodeBase + (long)g * N_EL) * COUT + col;
        { // mt = 0: rows 0..15, stores unconditional
            v16bf aA  = load_a(&At[l16 * 32], 0, hf8);
            const unsigned short* ar = xrow + l16 * CIN;
            v16bf ax0 = load_a(ar, 0, hf8), ax1 = load_a(ar, 32, hf8);
            v8f c = {bv, bv, bv, bv, bv, bv, bv, bv};
            c = WMMA(aA,  by,  c);
            c = WMMA(ax0, bw0, c);
            c = WMMA(ax1, bw1, c);
            #pragma unroll
            for (int r = 0; r < 8; ++r)
                __builtin_nontemporal_store(c[r], og + (hf8 + r) * COUT);
        }
        { // mt = 1: rows 16..18 only
            v16bf aA  = load_a(&At[(16 + l16) * 32], 0, hf8);
            const unsigned short* ar = xrow + (16 + l16) * CIN;
            v16bf ax0 = load_a(ar, 0, hf8), ax1 = load_a(ar, 32, hf8);
            v8f c = {bv, bv, bv, bv, bv, bv, bv, bv};
            c = WMMA(aA,  by,  c);
            c = WMMA(ax0, bw0, c);
            c = WMMA(ax1, bw1, c);
            if (hf == 0) {
                #pragma unroll
                for (int r = 0; r < 3; ++r)
                    __builtin_nontemporal_store(c[r], og + (16 + r) * COUT);
            }
        }
    }
}

extern "C" void kernel_launch(void* const* d_in, const int* in_sizes, int n_in,
                              void* d_out, int out_size, void* d_ws, size_t ws_size,
                              hipStream_t stream) {
    const float* x     = (const float*)d_in[0];
    const float* ew    = (const float*)d_in[1];
    const float* Wrel  = (const float*)d_in[2];
    const float* Wroot = (const float*)d_in[3];
    const float* bias  = (const float*)d_in[4];
    // d_in[5] = edge_index: dense per-graph 19x19 tiling is implicit; unused.

    int nNodes  = in_sizes[0] / CIN;      // 155648
    int nGraphs = nNodes / N_EL;          // 8192
    int blocks  = nGraphs / GPW;          // 2048

    graphconv_fused_kernel<<<blocks, 256, 0, stream>>>(x, ew, Wrel, Wroot, bias,
                                                       (float*)d_out);
}
